// AdaPTEncoder_33225867002014
// MI455X (gfx1250) — matrix operand order repaired and checked by reading, not compiled, not run on hardware
//
#include <hip/hip_runtime.h>

// ---------------- types & helpers ----------------
typedef __attribute__((ext_vector_type(16))) __bf16 v16bf;
typedef __attribute__((ext_vector_type(8)))  __bf16 v8bf;
typedef __attribute__((ext_vector_type(8)))  float  v8f;
typedef __attribute__((ext_vector_type(4)))  int    v4i;

#define AS1 __attribute__((address_space(1)))
#define AS3 __attribute__((address_space(3)))

// CDNA5 async global->LDS path (ASYNCcnt), guarded so the file compiles on any toolchain
#if defined(__has_builtin)
#if __has_builtin(__builtin_amdgcn_global_load_async_to_lds_b128) && __has_builtin(__builtin_amdgcn_s_wait_asynccnt)
#define GEMM_ASYNC 1
#endif
#endif
#ifndef GEMM_ASYNC
#define GEMM_ASYNC 0
#endif

union V16U { v16bf v; v8bf h[2]; };
__device__ __forceinline__ v16bf make16(v8bf lo, v8bf hi){ V16U u; u.h[0]=lo; u.h[1]=hi; return u.v; }

__device__ __forceinline__ float gelu_f(float x){ return 0.5f*x*(1.0f + erff(x*0.70710678118654752f)); }

#define BATCH 64
#define SEQ   197
#define SEQP  224          // padded key dim (14 tiles of 16)
#define CDIM  768
#define HEADS 12
#define DHEAD 64
#define MDIM  3072
#define TOK   (BATCH*SEQ)       // 12608
#define PTOK  (BATCH*(SEQ-1))   // 12544

// 256-thread block sum (8 wave32s)
__device__ __forceinline__ float bsum256(float v, float* red){
  #pragma unroll
  for (int d=1; d<32; d<<=1) v += __shfl_xor(v, d, 32);
  int tid = threadIdx.x;
  if ((tid & 31) == 0) red[tid>>5] = v;
  __syncthreads();
  float t = (tid < 8) ? red[tid] : 0.f;
  #pragma unroll
  for (int d=1; d<8; d<<=1) t += __shfl_xor(t, d, 32);
  if (tid == 0) red[0] = t;
  __syncthreads();
  float r = red[0];
  __syncthreads();
  return r;
}

// ---------------- LayerNorm -> bf16 (optional xt-gather) ----------------
__global__ __launch_bounds__(256) void ln_bf16_k(
    const float* __restrict__ X, const float* __restrict__ g, const float* __restrict__ bt,
    __bf16* __restrict__ out, int gather)
{
  __shared__ float red[16];
  int r = blockIdx.x;
  int src = gather ? ((r/196)*197 + 1 + (r%196)) : r;
  const float* xp = X + (size_t)src*CDIM;
  int tid = threadIdx.x;
  float lx[3]; float s = 0.f;
  #pragma unroll
  for (int u=0;u<3;u++){ lx[u] = xp[tid + u*256]; s += lx[u]; }
  float mu = bsum256(s, red) * (1.0f/768.0f);
  float v2 = 0.f;
  #pragma unroll
  for (int u=0;u<3;u++){ float d = lx[u]-mu; v2 += d*d; }
  float var = bsum256(v2, red) * (1.0f/768.0f);
  float inv = rsqrtf(var + 1e-5f);
  #pragma unroll
  for (int u=0;u<3;u++){
    int j = tid + u*256;
    out[(size_t)r*CDIM + j] = (__bf16)((lx[u]-mu)*inv*g[j] + bt[j]);
  }
}

// ---------------- weight convert+transpose: W[K][N] f32 -> Wt[N][K] bf16 ----------------
__global__ __launch_bounds__(256) void cvtT_k(const float* __restrict__ W, __bf16* __restrict__ Wt,
                                              int K, int N)
{
  size_t idx = (size_t)blockIdx.x*256 + threadIdx.x;
  if (idx >= (size_t)N*K) return;
  int n = (int)(idx / K), k = (int)(idx % K);
  Wt[idx] = (__bf16)W[(size_t)k*N + n];
}

// ---------------- tiled bf16 WMMA GEMM, double-buffered LDS ----------------
// C = A[M,K] * Wt[N,K]^T (+bias), modes:
// 0: f32 out  1: bf16 gelu out  2: bf16 out  3: QKV scatter  4: f32 out + residual
// OOB rows/cols handled by clamping load addresses (their products are never stored).
__global__ __launch_bounds__(256) void gemm_k(
    const __bf16* __restrict__ A, const __bf16* __restrict__ Wt,
    const float* __restrict__ bias, const float* __restrict__ resid,
    float* __restrict__ outF, __bf16* __restrict__ outB,
    __bf16* __restrict__ Qo, __bf16* __restrict__ Ko, __bf16* __restrict__ Vo,
    int Mtot, int Ntot, int Ktot, int mode)
{
  __shared__ __bf16 As[2][128*32];
  __shared__ __bf16 Bs[2][128*32];
  const int tid  = threadIdx.x;
  const int lane = tid & 31, wid = tid >> 5;
  const int l15  = lane & 15, hg = lane >> 4;
  const int m0 = blockIdx.y * 128, n0 = blockIdx.x * 128;
  const int wm = (wid >> 2) * 64, wn = (wid & 3) * 32;
  const int lr = tid >> 1, lseg = tid & 1;

  int gr = m0 + lr; if (gr > Mtot-1) gr = Mtot-1;
  int gn = n0 + lr; if (gn > Ntot-1) gn = Ntot-1;
  const __bf16* Ap = A  + (size_t)gr*Ktot + lseg*16;   // this thread's 32B slice of the A row
  const __bf16* Bp = Wt + (size_t)gn*Ktot + lseg*16;
  const int loff = lr*32 + lseg*16;

  auto stage = [&](int buf, int k0){
#if GEMM_ASYNC
    // 4 wave-wide async fills per stage; tracked by ASYNCcnt, LDS written directly
    __builtin_amdgcn_global_load_async_to_lds_b128((AS1 v4i*)(Ap + k0), (AS3 v4i*)(&As[buf][loff]), 0,  0);
    __builtin_amdgcn_global_load_async_to_lds_b128((AS1 v4i*)(Ap + k0), (AS3 v4i*)(&As[buf][loff]), 16, 0);
    __builtin_amdgcn_global_load_async_to_lds_b128((AS1 v4i*)(Bp + k0), (AS3 v4i*)(&Bs[buf][loff]), 0,  0);
    __builtin_amdgcn_global_load_async_to_lds_b128((AS1 v4i*)(Bp + k0), (AS3 v4i*)(&Bs[buf][loff]), 16, 0);
#else
    const uint4* sa = (const uint4*)(Ap + k0);
    uint4 a0 = sa[0], a1 = sa[1];
    const uint4* sb = (const uint4*)(Bp + k0);
    uint4 b0 = sb[0], b1 = sb[1];
    if (k0 + 32 < Ktot){
      __builtin_prefetch(Ap + k0 + 32, 0, 3);   // global_prefetch_b8, near scope
      __builtin_prefetch(Bp + k0 + 32, 0, 3);
    }
    uint4* da = (uint4*)(&As[buf][loff]); da[0] = a0; da[1] = a1;
    uint4* db = (uint4*)(&Bs[buf][loff]); db[0] = b0; db[1] = b1;
#endif
  };

  v8f zero8 = {0.f,0.f,0.f,0.f,0.f,0.f,0.f,0.f};
  v8f acc[4][2];
  #pragma unroll
  for (int a_=0;a_<4;a_++)
    #pragma unroll
    for (int b_=0;b_<2;b_++) acc[a_][b_] = zero8;

  stage(0, 0);
  int cur = 0;
  for (int k0 = 0; k0 < Ktot; k0 += 32){
    const int nxt = cur ^ 1;
    const bool has_next = (k0 + 32 < Ktot);
    if (has_next) stage(nxt, k0 + 32);          // overlap next fill with this compute
#if GEMM_ASYNC
    if (has_next) __builtin_amdgcn_s_wait_asynccnt(4);  // cur's 4 fills done; nxt's stay in flight
    else          __builtin_amdgcn_s_wait_asynccnt(0);
#endif
    __syncthreads();

    const __bf16* Asb = As[cur];
    const __bf16* Bsb = Bs[cur];
    v16bf af[4], bfm[2];
    #pragma unroll
    for (int ti=0;ti<4;ti++){
      int row = wm + ti*16 + l15;
      v8bf lo = *(const v8bf*)(Asb + row*32 + hg*16);
      v8bf hi = *(const v8bf*)(Asb + row*32 + hg*16 + 8);
      af[ti] = make16(lo,hi);
    }
    #pragma unroll
    for (int tj=0;tj<2;tj++){
      int nr = wn + tj*16 + l15;
      v8bf lo = *(const v8bf*)(Bsb + nr*32 + hg*16);
      v8bf hi = *(const v8bf*)(Bsb + nr*32 + hg*16 + 8);
      bfm[tj] = make16(lo,hi);
    }
    #pragma unroll
    for (int ti=0;ti<4;ti++)
      #pragma unroll
      for (int tj=0;tj<2;tj++)
        acc[ti][tj] = __builtin_amdgcn_wmma_f32_16x16x32_bf16(
            false, af[ti], false, bfm[tj], (short)0, acc[ti][tj], false, false);
    __syncthreads();
    cur = nxt;
  }

  #pragma unroll
  for (int ti=0;ti<4;ti++){
    #pragma unroll
    for (int tj=0;tj<2;tj++){
      int col = n0 + wn + tj*16 + l15;
      if (col >= Ntot) continue;
      float bv = bias ? bias[col] : 0.f;
      #pragma unroll
      for (int i=0;i<8;i++){
        int row = m0 + wm + ti*16 + i + hg*8;
        if (row >= Mtot) continue;
        float v = acc[ti][tj][i] + bv;
        if (mode == 4){
          outF[(size_t)row*Ntot + col] = v + resid[(size_t)row*Ntot + col];
        } else if (mode == 1){
          outB[(size_t)row*Ntot + col] = (__bf16)gelu_f(v);
        } else if (mode == 2){
          outB[(size_t)row*Ntot + col] = (__bf16)v;
        } else if (mode == 0){
          outF[(size_t)row*Ntot + col] = v;
        } else { // mode 3: scatter QKV to per-head layouts (V transposed + padded)
          int which = col / CDIM;
          int hc = col - which*CDIM;
          int hh = hc >> 6, dh = hc & 63;
          int b  = row / SEQ, n = row - b*SEQ;
          size_t bh = (size_t)(b*HEADS + hh);
          __bf16 val = (__bf16)v;
          if (which == 0)      Qo[(bh*SEQ + n)*DHEAD + dh] = val;
          else if (which == 1) Ko[(bh*SEQ + n)*DHEAD + dh] = val;
          else                 Vo[(bh*DHEAD + dh)*SEQP + n] = val;
        }
      }
    }
  }
}

// ---------------- fused masked attention (per b,h), softmax in registers ----------------
__global__ __launch_bounds__(256) void attn_wmma(
    const __bf16* __restrict__ Qb, const __bf16* __restrict__ Kb,
    const __bf16* __restrict__ Vt, const float* __restrict__ dvec,
    int use_mask, __bf16* __restrict__ O)
{
  __shared__ __bf16 Pl[8][16][SEQP];  // per-wave P staging (C-layout -> A-layout)
  const int tid = threadIdx.x;
  const int lane = tid & 31, wid = tid >> 5;
  const int l15 = lane & 15, hg = lane >> 4;
  const int bh = blockIdx.y;
  const int b = bh / HEADS, h = bh % HEADS;
  const int qrow0 = blockIdx.x*128 + wid*16;
  const __bf16* Qp = Qb + (size_t)bh*SEQ*DHEAD;
  const __bf16* Kp = Kb + (size_t)bh*SEQ*DHEAD;
  const __bf16* Vp = Vt + (size_t)bh*DHEAD*SEQP;

  int qr = qrow0 + l15; if (qr > SEQ-1) qr = SEQ-1;
  v16bf qa[2];
  #pragma unroll
  for (int kc=0;kc<2;kc++){
    int kb0 = kc*32 + hg*16;
    v8bf lo = *(const v8bf*)(Qp + qr*DHEAD + kb0);
    v8bf hi = *(const v8bf*)(Qp + qr*DHEAD + kb0 + 8);
    qa[kc] = make16(lo,hi);
  }

  v8f zero8 = {0.f,0.f,0.f,0.f,0.f,0.f,0.f,0.f};
  float sc[13][8];
  #pragma unroll
  for (int t=0;t<13;t++){
    int kr = t*16 + l15; if (kr > SEQ-1) kr = SEQ-1;
    v8f s = zero8;
    #pragma unroll
    for (int kc=0;kc<2;kc++){
      int ko = kc*32 + hg*16;
      v8bf lo = *(const v8bf*)(Kp + kr*DHEAD + ko);
      v8bf hi = *(const v8bf*)(Kp + kr*DHEAD + ko + 8);
      v16bf kf = make16(lo,hi);
      s = __builtin_amdgcn_wmma_f32_16x16x32_bf16(false, qa[kc], false, kf, (short)0, s, false, false);
    }
    #pragma unroll
    for (int i=0;i<8;i++){
      int qi = qrow0 + i + hg*8;
      int kj = t*16 + l15;
      float v = s[i]*0.125f;                 // 1/sqrt(64)
      bool ok = (qi < SEQ) && (kj < SEQ);
      if (ok && use_mask && qi != kj) ok = (dvec[b*SEQ+qi]*dvec[b*SEQ+kj] > 0.f);
      sc[t][i] = ok ? v : -1.0e9f;
    }
  }
  // masked softmax over 208 keys: reduce across 13 tiles + 16 lanes
  #pragma unroll
  for (int i=0;i<8;i++){
    float m = sc[0][i];
    #pragma unroll
    for (int t=1;t<13;t++) m = fmaxf(m, sc[t][i]);
    #pragma unroll
    for (int d=1; d<16; d<<=1) m = fmaxf(m, __shfl_xor(m, d, 32));
    float sum = 0.f;
    #pragma unroll
    for (int t=0;t<13;t++){ float e = __expf(sc[t][i]-m); sc[t][i]=e; sum += e; }
    #pragma unroll
    for (int d=1; d<16; d<<=1) sum += __shfl_xor(sum, d, 32);
    float inv = 1.f/sum;
    #pragma unroll
    for (int t=0;t<13;t++) sc[t][i] *= inv;
  }
  // stage P (zero-pad tile 13)
  #pragma unroll
  for (int t=0;t<14;t++)
    #pragma unroll
    for (int i=0;i<8;i++)
      Pl[wid][i + hg*8][t*16 + l15] = (t<13) ? (__bf16)sc[t][i] : (__bf16)0.f;
  __syncthreads();

  v8f oa[4];
  #pragma unroll
  for (int tj=0;tj<4;tj++) oa[tj] = zero8;
  #pragma unroll
  for (int ks=0;ks<7;ks++){
    int kb0 = ks*32 + hg*16;
    v8bf lo = *(const v8bf*)(&Pl[wid][l15][kb0]);
    v8bf hi = *(const v8bf*)(&Pl[wid][l15][kb0+8]);
    v16bf pf = make16(lo,hi);
    int ko = ks*32 + hg*16;
    #pragma unroll
    for (int tj=0;tj<4;tj++){
      int dh = tj*16 + l15;
      v8bf vl = *(const v8bf*)(Vp + dh*SEQP + ko);
      v8bf vh = *(const v8bf*)(Vp + dh*SEQP + ko + 8);
      v16bf vf = make16(vl,vh);
      oa[tj] = __builtin_amdgcn_wmma_f32_16x16x32_bf16(false, pf, false, vf, (short)0, oa[tj], false, false);
    }
  }
  #pragma unroll
  for (int tj=0;tj<4;tj++){
    int dh = tj*16 + l15;
    #pragma unroll
    for (int i=0;i<8;i++){
      int row = qrow0 + i + hg*8;
      if (row < SEQ)
        O[((size_t)b*SEQ + row)*CDIM + h*DHEAD + dh] = (__bf16)oa[tj][i];
    }
  }
}

// ---------------- predictor helpers ----------------
__global__ __launch_bounds__(256) void pool_k(const __bf16* __restrict__ ph,
                                              const float* __restrict__ prev,
                                              float* __restrict__ glob)
{
  __shared__ float red[16];
  __shared__ float dsh;
  int b = blockIdx.x, tid = threadIdx.x;
  float s = (tid < 196) ? prev[b*196 + tid] : 0.f;
  float den = bsum256(s, red);
  if (tid==0) dsh = den + 1e-20f;
  __syncthreads();
  float dn = dsh;
  for (int j = tid; j < 384; j += 256){
    float acc = 0.f;
    for (int n=0;n<196;n++)
      acc += (float)ph[((size_t)b*196 + n)*CDIM + 384 + j] * prev[b*196+n];
    glob[b*384 + j] = acc / dn;
  }
}

__global__ __launch_bounds__(256) void concat_k(const __bf16* __restrict__ ph,
                                                const float* __restrict__ glob,
                                                __bf16* __restrict__ hp)
{
  size_t idx = (size_t)blockIdx.x*256 + threadIdx.x;
  if (idx >= (size_t)PTOK*CDIM) return;
  int j = (int)(idx % CDIM); size_t r = idx / CDIM; int b = (int)(r/196);
  hp[idx] = (j < 384) ? ph[r*CDIM + j] : (__bf16)glob[b*384 + (j-384)];
}

__device__ __forceinline__ float u01h(unsigned long long z){
  z += 0x9E3779B97F4A7C15ull;
  z = (z ^ (z>>30)) * 0xBF58476D1CE4E5B9ull;
  z = (z ^ (z>>27)) * 0x94D049BB133111EBull;
  z ^= (z>>31);
  return ((float)(z >> 40) + 0.5f) * (1.0f/16777216.0f);
}

__global__ __launch_bounds__(256) void head_k(const __bf16* __restrict__ h3,
                                              const float* __restrict__ w3,
                                              const float* __restrict__ b3,
                                              float* __restrict__ prev,
                                              float* __restrict__ dvec, int pi)
{
  int r = blockIdx.x*256 + threadIdx.x;
  if (r >= PTOK) return;
  float l0 = b3[0], l1 = b3[1];
  for (int k=0;k<192;k++){
    float hv = (float)h3[(size_t)r*192 + k];
    l0 += hv * w3[k*2+0];
    l1 += hv * w3[k*2+1];
  }
  float mx = fmaxf(l0,l1);
  float e0 = __expf(l0-mx), e1 = __expf(l1-mx);
  float inv = 1.f/(e0+e1);
  float s0 = e0*inv, s1 = e1*inv;
  unsigned long long base = ((unsigned long long)(pi+1) << 32) ^ ((unsigned long long)r*2ull);
  float u0 = fmaxf(u01h(base),   1e-10f);
  float u1 = fmaxf(u01h(base+1), 1e-10f);
  float g0 = -logf(-logf(u0));
  float g1 = -logf(-logf(u1));
  float hard1 = ((s1+g1) > (s0+g0)) ? 1.f : 0.f;  // argmax(softmax(soft+g)) == argmax(soft+g)
  float dec = hard1 * prev[r];
  prev[r] = dec;
  int b = r/196, n = r%196;
  dvec[b*SEQ + 1 + n] = dec;
  if (n==0) dvec[b*SEQ] = 1.f;   // CLS always kept
}

__global__ __launch_bounds__(256) void init_k(float* prev, float* dvec){
  int r = blockIdx.x*256 + threadIdx.x;
  if (r < PTOK) prev[r] = 1.f;
  if (r < TOK)  dvec[r] = 1.f;
}

// ---------------- host orchestration ----------------
extern "C" void kernel_launch(void* const* d_in, const int* in_sizes, int n_in,
                              void* d_out, int out_size, void* d_ws, size_t ws_size,
                              hipStream_t stream)
{
  const float* in_x   = (const float*)d_in[0];
  const float* ln1_g  = (const float*)d_in[1];
  const float* ln1_b  = (const float*)d_in[2];
  const float* qkv_w  = (const float*)d_in[3];
  const float* qkv_b  = (const float*)d_in[4];
  const float* proj_w = (const float*)d_in[5];
  const float* proj_b = (const float*)d_in[6];
  const float* ln2_g  = (const float*)d_in[7];
  const float* ln2_b  = (const float*)d_in[8];
  const float* mlp_w1 = (const float*)d_in[9];
  const float* mlp_b1 = (const float*)d_in[10];
  const float* mlp_w2 = (const float*)d_in[11];
  const float* mlp_b2 = (const float*)d_in[12];
  const float* p_ln_g = (const float*)d_in[13];
  const float* p_ln_b = (const float*)d_in[14];
  const float* p_in_w = (const float*)d_in[15];
  const float* p_in_b = (const float*)d_in[16];
  const float* p_w1   = (const float*)d_in[17];
  const float* p_b1   = (const float*)d_in[18];
  const float* p_w2   = (const float*)d_in[19];
  const float* p_b2   = (const float*)d_in[20];
  const float* p_w3   = (const float*)d_in[21];
  const float* p_b3   = (const float*)d_in[22];
  (void)in_sizes; (void)n_in; (void)out_size;

  char* ws = (char*)d_ws; size_t off = 0;
  auto alloc = [&](size_t bytes)->char*{
    char* p = ws + off; off += (bytes + 255) & ~(size_t)255; return p; };

  float*  X     = (float*) alloc((size_t)TOK*CDIM*4);
  float*  X2    = (float*) alloc((size_t)TOK*CDIM*4);
  __bf16* Abuf  = (__bf16*)alloc((size_t)TOK*CDIM*2);
  __bf16* H1    = (__bf16*)alloc((size_t)TOK*MDIM*2);
  __bf16* Qb    = (__bf16*)alloc((size_t)BATCH*HEADS*SEQ*DHEAD*2);
  __bf16* Kb    = (__bf16*)alloc((size_t)BATCH*HEADS*SEQ*DHEAD*2);
  __bf16* Vt    = (__bf16*)alloc((size_t)BATCH*HEADS*DHEAD*SEQP*2);
  __bf16* Ob    = (__bf16*)alloc((size_t)TOK*CDIM*2);
  __bf16* wt_qkv= (__bf16*)alloc((size_t)3*CDIM*CDIM*2);
  __bf16* wt_prj= (__bf16*)alloc((size_t)CDIM*CDIM*2);
  __bf16* wt_m1 = (__bf16*)alloc((size_t)CDIM*MDIM*2);
  __bf16* wt_m2 = (__bf16*)alloc((size_t)CDIM*MDIM*2);
  __bf16* wt_pin= (__bf16*)alloc((size_t)CDIM*CDIM*2);
  __bf16* wt_p1 = (__bf16*)alloc((size_t)CDIM*(CDIM/2)*2);
  __bf16* wt_p2 = (__bf16*)alloc((size_t)(CDIM/2)*(CDIM/4)*2);
  float*  glob  = (float*) alloc((size_t)BATCH*(CDIM/2)*4);
  float*  prev  = (float*) alloc((size_t)PTOK*4);
  float*  dvec  = (float*) alloc((size_t)TOK*4);
  if (off > ws_size) return;

  // predictor scratch carved from H1 (free at predictor time)
  __bf16* p_h  = H1;
  __bf16* p_hp = H1 + (size_t)PTOK*CDIM;
  __bf16* p_h2 = H1 + (size_t)2*PTOK*CDIM;
  __bf16* p_h3 = p_h2 + (size_t)PTOK*(CDIM/2);

  (void)hipMemcpyAsync(X, in_x, (size_t)TOK*CDIM*4, hipMemcpyDeviceToDevice, stream);
  (void)hipMemsetAsync(Vt, 0, (size_t)BATCH*HEADS*DHEAD*SEQP*2, stream);  // zero key-padding
  init_k<<<(TOK+255)/256, 256, 0, stream>>>(prev, dvec);

  int p = 0;
  for (int i = 0; i < 12; i++){
    if (i==3 || i==6 || i==9){
      ln_bf16_k<<<PTOK, 256, 0, stream>>>(X, p_ln_g + p*CDIM, p_ln_b + p*CDIM, Abuf, 1);
      cvtT_k<<<(int)(((size_t)CDIM*CDIM+255)/256), 256, 0, stream>>>(p_in_w + (size_t)p*CDIM*CDIM, wt_pin, CDIM, CDIM);
      gemm_k<<<dim3(CDIM/128, PTOK/128), 256, 0, stream>>>(Abuf, wt_pin, p_in_b + p*CDIM, nullptr,
          nullptr, p_h, nullptr, nullptr, nullptr, PTOK, CDIM, CDIM, 1);
      pool_k<<<BATCH, 256, 0, stream>>>(p_h, prev, glob);
      concat_k<<<(int)(((size_t)PTOK*CDIM+255)/256), 256, 0, stream>>>(p_h, glob, p_hp);
      cvtT_k<<<(int)(((size_t)384*CDIM+255)/256), 256, 0, stream>>>(p_w1 + (size_t)p*CDIM*384, wt_p1, CDIM, 384);
      gemm_k<<<dim3(3, PTOK/128), 256, 0, stream>>>(p_hp, wt_p1, p_b1 + p*384, nullptr,
          nullptr, p_h2, nullptr, nullptr, nullptr, PTOK, 384, CDIM, 1);
      cvtT_k<<<(int)(((size_t)192*384+255)/256), 256, 0, stream>>>(p_w2 + (size_t)p*384*192, wt_p2, 384, 192);
      gemm_k<<<dim3(2, PTOK/128), 256, 0, stream>>>(p_h2, wt_p2, p_b2 + p*192, nullptr,
          nullptr, p_h3, nullptr, nullptr, nullptr, PTOK, 192, 384, 1);
      head_k<<<(PTOK+255)/256, 256, 0, stream>>>(p_h3, p_w3 + (size_t)p*192*2, p_b3 + p*2, prev, dvec, p);
      p++;
    }
    // --- attention block ---
    ln_bf16_k<<<TOK, 256, 0, stream>>>(X, ln1_g + i*CDIM, ln1_b + i*CDIM, Abuf, 0);
    cvtT_k<<<(int)(((size_t)3*CDIM*CDIM+255)/256), 256, 0, stream>>>(qkv_w + (size_t)i*CDIM*3*CDIM, wt_qkv, CDIM, 3*CDIM);
    gemm_k<<<dim3(3*CDIM/128, (TOK+127)/128), 256, 0, stream>>>(Abuf, wt_qkv, qkv_b + (size_t)i*3*CDIM, nullptr,
        nullptr, nullptr, Qb, Kb, Vt, TOK, 3*CDIM, CDIM, 3);
    attn_wmma<<<dim3(2, BATCH*HEADS), 256, 0, stream>>>(Qb, Kb, Vt, dvec, (p>0)?1:0, Ob);
    cvtT_k<<<(int)(((size_t)CDIM*CDIM+255)/256), 256, 0, stream>>>(proj_w + (size_t)i*CDIM*CDIM, wt_prj, CDIM, CDIM);
    gemm_k<<<dim3(CDIM/128, (TOK+127)/128), 256, 0, stream>>>(Ob, wt_prj, proj_b + i*CDIM, X,
        X2, nullptr, nullptr, nullptr, nullptr, TOK, CDIM, CDIM, 4);
    // --- MLP block ---
    ln_bf16_k<<<TOK, 256, 0, stream>>>(X2, ln2_g + i*CDIM, ln2_b + i*CDIM, Abuf, 0);
    cvtT_k<<<(int)(((size_t)CDIM*MDIM+255)/256), 256, 0, stream>>>(mlp_w1 + (size_t)i*CDIM*MDIM, wt_m1, CDIM, MDIM);
    gemm_k<<<dim3(MDIM/128, (TOK+127)/128), 256, 0, stream>>>(Abuf, wt_m1, mlp_b1 + (size_t)i*MDIM, nullptr,
        nullptr, H1, nullptr, nullptr, nullptr, TOK, MDIM, CDIM, 1);
    cvtT_k<<<(int)(((size_t)CDIM*MDIM+255)/256), 256, 0, stream>>>(mlp_w2 + (size_t)i*MDIM*CDIM, wt_m2, MDIM, CDIM);
    float* outX = (i==11) ? (float*)d_out : X;
    gemm_k<<<dim3(CDIM/128, (TOK+127)/128), 256, 0, stream>>>(H1, wt_m2, mlp_b2 + i*CDIM, X2,
        outX, nullptr, nullptr, nullptr, nullptr, TOK, CDIM, MDIM, 4);
  }
}
